// AdvancedEmbeddingBlock_15479062134840
// MI455X (gfx1250) — matrix operand order repaired and checked by reading, not compiled, not run on hardware
//
#include <hip/hip_runtime.h>
#include <hip/hip_bf16.h>

// ---------------------------------------------------------------------------
// Types
// ---------------------------------------------------------------------------
typedef __attribute__((ext_vector_type(16))) __bf16 v16bf;
typedef __attribute__((ext_vector_type(8)))  __bf16 bf16x8;
typedef __attribute__((ext_vector_type(8)))  float  v8f;
typedef __attribute__((ext_vector_type(4)))  float  f32x4;

#define HID      256
#define RAD      64
#define NTILES   16          // 256 / 16 output column tiles
#define KB_FEAT  8           // 256 / 32 K-blocks per feature operand
#define KB_RBF   2           // 64 / 32
#define KB_TOT   24          // 768 / 32 for lin_w
#define WAVES_PER_BLOCK 4
#define LDS_STRIDE (HID + 8) // pad to dodge bank conflicts

// ---------------------------------------------------------------------------
// Prep kernel 1: per-node feature vector -> bf16 [N_NODES, 256]
// layout: [ emb(128) | tag(32) | phys(32) | period(32) | group(32) ]
// ---------------------------------------------------------------------------
__global__ void node_feat_kernel(const int* __restrict__ x,
                                 const int* __restrict__ tag,
                                 const int* __restrict__ period_idx,
                                 const int* __restrict__ group_idx,
                                 const float* __restrict__ props,      // [85,18]
                                 const float* __restrict__ emb_w,      // [85,128]
                                 const float* __restrict__ tag_w,      // [3,32]
                                 const float* __restrict__ period_w,   // [8,32]
                                 const float* __restrict__ group_w,    // [19,32]
                                 const float* __restrict__ phys_w,     // [18,32]
                                 const float* __restrict__ phys_b,     // [32]
                                 __bf16* __restrict__ xfeat)           // [N,256]
{
    const int node = blockIdx.x;
    const int col  = threadIdx.x;
    const int z    = x[node];
    float v;
    if (col < 128) {
        v = emb_w[z * 128 + col];
    } else if (col < 160) {
        v = tag_w[tag[node] * 32 + (col - 128)];
    } else if (col < 192) {
        const int c = col - 160;
        float s = phys_b[c];
        #pragma unroll
        for (int p = 0; p < 18; ++p)
            s += props[z * 18 + p] * phys_w[p * 32 + c];
        v = s;
    } else if (col < 224) {
        v = period_w[period_idx[z] * 32 + (col - 192)];
    } else {
        v = group_w[group_idx[z] * 32 + (col - 224)];
    }
    xfeat[(long)node * HID + col] = (__bf16)v;
}

// ---------------------------------------------------------------------------
// Prep kernel 2: swizzle a row-major f32 weight [K, 256] into per-lane WMMA
// B-fragment layout (bf16). Tile order: [n_tile][k_block][lane(32)][16].
// Lane l of tile (n,k): column n*16 + (l&15), K = k*32 + (l>>4)*8 + (t&7) + (t>>3)*16.
// ---------------------------------------------------------------------------
__global__ void swizzle_w_kernel(const float* __restrict__ W,
                                 __bf16* __restrict__ out,
                                 int Kdim, int total)
{
    const int tid = blockIdx.x * blockDim.x + threadIdx.x;
    if (tid >= total) return;
    const int kblocks = Kdim >> 5;
    const int t    = tid & 15;
    const int lane = (tid >> 4) & 31;
    const int kb   = (tid >> 9) % kblocks;
    const int nt   = tid / (512 * kblocks);
    const int k = kb * 32 + (lane >> 4) * 8 + (t & 7) + (t >> 3) * 16;
    const int n = nt * 16 + (lane & 15);
    out[tid] = (__bf16)W[(long)k * HID + n];
}

// ---------------------------------------------------------------------------
// Fragment helpers
// ---------------------------------------------------------------------------
__device__ __forceinline__ v16bf make_afrag(bf16x8 lo, bf16x8 hi) {
    return __builtin_shufflevector(lo, hi, 0, 1, 2, 3, 4, 5, 6, 7,
                                           8, 9, 10, 11, 12, 13, 14, 15);
}

// A fragment from a row-major bf16 row (stride 256), K-block kb
__device__ __forceinline__ v16bf load_afrag_global(const __bf16* rowbase,
                                                   int kb, int khalf) {
    const bf16x8* p = (const bf16x8*)rowbase;
    bf16x8 lo = p[kb * 4 + khalf];       // K = kb*32 + khalf*8 .. +7
    bf16x8 hi = p[kb * 4 + khalf + 2];   // K = +16 .. +23
    return make_afrag(lo, hi);
}

// B fragment from pre-swizzled buffer (32 contiguous bytes per lane)
__device__ __forceinline__ v16bf load_bfrag(const __bf16* swz, int ntile,
                                            int kglob, int kbtot, int lane) {
    return *(const v16bf*)(swz + (((long)(ntile * kbtot + kglob)) * 32 + lane) * 16);
}

// swish with HW reciprocal: v_exp_f32 + v_add + v_rcp_f32 + v_mul
// (avoids the correctly-rounded division expansion)
__device__ __forceinline__ float swish_f(float v) {
    return v * __builtin_amdgcn_rcpf(1.0f + __expf(-v));
}

// ---------------------------------------------------------------------------
// Main fused edge kernel: out = swish( xi@W0 + xj@W1 + swish(rbf@Wr+br)@W2 + b )
// 1 wave (32 lanes) = 16 edges x 256 cols. 4 waves / block.
// ---------------------------------------------------------------------------
__global__ __launch_bounds__(WAVES_PER_BLOCK * 32)
void edge_gemm_kernel(const int*   __restrict__ idx_i,
                      const int*   __restrict__ idx_j,
                      const float* __restrict__ rbf,        // [E,64] f32
                      const __bf16* __restrict__ xfeat,     // [N,256] bf16
                      const __bf16* __restrict__ rbfw_swz,  // [16][2][32][16]
                      const __bf16* __restrict__ linw_swz,  // [16][24][32][16]
                      const float* __restrict__ rbf_b,      // [256]
                      const float* __restrict__ lin_b,      // [256]
                      float* __restrict__ out,              // [E,256]
                      int n_edges)
{
    __shared__ __align__(16) __bf16 lds_h[WAVES_PER_BLOCK][16][LDS_STRIDE];

    const int lane  = threadIdx.x & 31;
    const int wave  = threadIdx.x >> 5;
    const int tile  = blockIdx.x * WAVES_PER_BLOCK + wave;
    if (tile * 16 >= n_edges) return;        // uniform per wave -> EXEC stays all-1

    const int row   = lane & 15;             // M-row handled by this lane
    const int khalf = lane >> 4;             // 0 or 1 (K-halving of A fragment)
    const long e    = (long)tile * 16 + row; // edge id for this lane's row

    const int ni = idx_i[e];
    const int nj = idx_j[e];

    // ---- A fragments for x_[i], x_[j] (kept in VGPRs for the whole N loop)
    v16bf ai[KB_FEAT], aj[KB_FEAT];
    {
        const __bf16* ri = xfeat + (long)ni * HID;
        const __bf16* rj = xfeat + (long)nj * HID;
        #pragma unroll
        for (int kb = 0; kb < KB_FEAT; ++kb) {
            ai[kb] = load_afrag_global(ri, kb, khalf);
            aj[kb] = load_afrag_global(rj, kb, khalf);
        }
    }

    // ---- A fragments for rbf (streamed once: non-temporal f32 loads -> bf16)
    v16bf ar[KB_RBF];
    {
        const float* rrow = rbf + e * RAD;
        #pragma unroll
        for (int kb = 0; kb < KB_RBF; ++kb) {
            const f32x4* lo = (const f32x4*)(rrow + kb * 32 + khalf * 8);
            f32x4 f0 = __builtin_nontemporal_load(lo + 0);
            f32x4 f1 = __builtin_nontemporal_load(lo + 1);
            f32x4 f2 = __builtin_nontemporal_load(lo + 4);   // +16 floats
            f32x4 f3 = __builtin_nontemporal_load(lo + 5);
            bf16x8 l, h;
            #pragma unroll
            for (int t = 0; t < 4; ++t) {
                l[t]     = (__bf16)f0[t];
                l[t + 4] = (__bf16)f1[t];
                h[t]     = (__bf16)f2[t];
                h[t + 4] = (__bf16)f3[t];
            }
            ar[kb] = make_afrag(l, h);
        }
    }

    // ---- rbf_h tile: 16x256, computed per N-tile, swished, staged to LDS
    for (int n = 0; n < NTILES; ++n) {
        const float bb = rbf_b[n * 16 + row];
        v8f h = {bb, bb, bb, bb, bb, bb, bb, bb};
        #pragma unroll
        for (int kb = 0; kb < KB_RBF; ++kb) {
            v16bf b = load_bfrag(rbfw_swz, n, kb, KB_RBF, lane);
            h = __builtin_amdgcn_wmma_f32_16x16x32_bf16(false, ar[kb], false, b,
                                                        (short)0, h, false, false);
        }
        #pragma unroll
        for (int v = 0; v < 8; ++v) {
            // C layout: VGPR v -> M = v + khalf*8, N = n*16 + row
            lds_h[wave][khalf * 8 + v][n * 16 + row] = (__bf16)swish_f(h[v]);
        }
    }

    // ---- reload rbf_h as A fragments from LDS (same-wave DS ops are in order)
    v16bf ah[KB_FEAT];
    #pragma unroll
    for (int kb = 0; kb < KB_FEAT; ++kb) {
        const __bf16* p = &lds_h[wave][row][kb * 32 + khalf * 8];
        bf16x8 lo = *(const bf16x8*)p;
        bf16x8 hi = *(const bf16x8*)(p + 16);
        ah[kb] = make_afrag(lo, hi);
    }

    // ---- main GEMM: 16 N-tiles x 24 K-blocks (xi | xj | rbf_h)
    for (int n = 0; n < NTILES; ++n) {
        const float bb = lin_b[n * 16 + row];
        v8f acc = {bb, bb, bb, bb, bb, bb, bb, bb};
        #pragma unroll
        for (int kb = 0; kb < KB_FEAT; ++kb) {
            v16bf b = load_bfrag(linw_swz, n, kb, KB_TOT, lane);
            acc = __builtin_amdgcn_wmma_f32_16x16x32_bf16(false, ai[kb], false, b,
                                                          (short)0, acc, false, false);
        }
        #pragma unroll
        for (int kb = 0; kb < KB_FEAT; ++kb) {
            v16bf b = load_bfrag(linw_swz, n, kb + 8, KB_TOT, lane);
            acc = __builtin_amdgcn_wmma_f32_16x16x32_bf16(false, aj[kb], false, b,
                                                          (short)0, acc, false, false);
        }
        #pragma unroll
        for (int kb = 0; kb < KB_FEAT; ++kb) {
            v16bf b = load_bfrag(linw_swz, n, kb + 16, KB_TOT, lane);
            acc = __builtin_amdgcn_wmma_f32_16x16x32_bf16(false, ah[kb], false, b,
                                                          (short)0, acc, false, false);
        }
        // store: edge = tile*16 + khalf*8 + v, col = n*16 + row
        // coalesced b32 stores; non-temporal so 410MB of output doesn't evict
        // the L2-resident node-feature table / weights.
        float* orow = out + ((long)tile * 16 + khalf * 8) * HID + n * 16 + row;
        #pragma unroll
        for (int v = 0; v < 8; ++v) {
            __builtin_nontemporal_store(swish_f(acc[v]), orow + (long)v * HID);
        }
    }
}

// ---------------------------------------------------------------------------
// Host-side launcher
// ---------------------------------------------------------------------------
extern "C" void kernel_launch(void* const* d_in, const int* in_sizes, int n_in,
                              void* d_out, int out_size, void* d_ws, size_t ws_size,
                              hipStream_t stream) {
    const int*   x          = (const int*)  d_in[0];
    const int*   tag        = (const int*)  d_in[1];
    const int*   idx_i      = (const int*)  d_in[2];
    const int*   idx_j      = (const int*)  d_in[3];
    const float* rbf        = (const float*)d_in[4];
    const int*   period_idx = (const int*)  d_in[5];
    const int*   group_idx  = (const int*)  d_in[6];
    const float* props      = (const float*)d_in[7];
    const float* emb_w      = (const float*)d_in[8];
    const float* tag_w      = (const float*)d_in[9];
    const float* period_w   = (const float*)d_in[10];
    const float* group_w    = (const float*)d_in[11];
    const float* phys_w     = (const float*)d_in[12];
    const float* phys_b     = (const float*)d_in[13];
    const float* rbf_w      = (const float*)d_in[14];
    const float* rbf_b      = (const float*)d_in[15];
    const float* lin_w      = (const float*)d_in[16];
    const float* lin_b      = (const float*)d_in[17];
    float*       out        = (float*)d_out;

    const int n_nodes = in_sizes[0];
    const int n_edges = in_sizes[2];

    // workspace carving (all offsets 256B-aligned)
    char* ws = (char*)d_ws;
    __bf16* xfeat    = (__bf16*)(ws);                                   // 25000*256*2 = 12.8 MB
    size_t  off1     = (size_t)n_nodes * HID * sizeof(__bf16);
    off1             = (off1 + 255) & ~(size_t)255;
    __bf16* rbfw_swz = (__bf16*)(ws + off1);                            // 64*256*2  = 32 KB
    size_t  off2     = off1 + (size_t)RAD * HID * sizeof(__bf16);
    off2             = (off2 + 255) & ~(size_t)255;
    __bf16* linw_swz = (__bf16*)(ws + off2);                            // 768*256*2 = 384 KB

    // 1) node features (bf16)
    node_feat_kernel<<<n_nodes, HID, 0, stream>>>(
        x, tag, period_idx, group_idx, props, emb_w, tag_w,
        period_w, group_w, phys_w, phys_b, xfeat);

    // 2) weight swizzles into WMMA B-fragment layout
    {
        int total_r = RAD * HID;                 // 16384
        swizzle_w_kernel<<<(total_r + 255) / 256, 256, 0, stream>>>(
            rbf_w, rbfw_swz, RAD, total_r);
        int total_l = 3 * HID * HID;             // 196608
        swizzle_w_kernel<<<(total_l + 255) / 256, 256, 0, stream>>>(
            lin_w, linw_swz, 3 * HID, total_l);
    }

    // 3) fused edge GEMM (1 wave = 16 edges; 4 waves per block)
    {
        int n_tiles  = (n_edges + 15) / 16;
        int n_blocks = (n_tiles + WAVES_PER_BLOCK - 1) / WAVES_PER_BLOCK;
        edge_gemm_kernel<<<n_blocks, WAVES_PER_BLOCK * 32, 0, stream>>>(
            idx_i, idx_j, rbf, xfeat, rbfw_swz, linw_swz,
            rbf_b, lin_b, out, n_edges);
    }
}
